// SelfAttention_14207751815795
// MI455X (gfx1250) — matrix operand order repaired and checked by reading, not compiled, not run on hardware
//
#include <hip/hip_runtime.h>
#include <hip/hip_bf16.h>

// B=2, S=2048, D=1024, H=16, HD=64
#define SQ   2048
#define DIM  1024
#define NH   16
#define HD   64
#define ROWS 4096          // B*S
#define D3   3072

typedef __attribute__((ext_vector_type(16))) __bf16 v16bf;
typedef __attribute__((ext_vector_type(8)))  __bf16 v8bf;
typedef __attribute__((ext_vector_type(8)))  float  v8f;

__device__ __forceinline__ unsigned short bfbits(float f) {
    unsigned u = __builtin_bit_cast(unsigned, f);
    return (unsigned short)((u + 0x7FFFu + ((u >> 16) & 1u)) >> 16);
}
__device__ __forceinline__ __bf16 f2bf(float f) {
    return __builtin_bit_cast(__bf16, bfbits(f));
}

__device__ __forceinline__ v16bf mk16(v8bf lo, v8bf hi) {
    v16bf r;
#pragma unroll
    for (int i = 0; i < 8; ++i) { r[i] = lo[i]; r[i + 8] = hi[i]; }
    return r;
}

__device__ __forceinline__ v8f wmma_bf16(v16bf a, v16bf b, v8f c) {
    return __builtin_amdgcn_wmma_f32_16x16x32_bf16(
        false, a, false, b, (short)0, c, false, false);
}

__device__ __forceinline__ float rmax16(float v) {
#pragma unroll
    for (int m = 1; m <= 8; m <<= 1) v = fmaxf(v, __shfl_xor(v, m, 16));
    return v;
}
__device__ __forceinline__ float rsum16(float v) {
#pragma unroll
    for (int m = 1; m <= 8; m <<= 1) v += __shfl_xor(v, m, 16);
    return v;
}

// ---------------------------------------------------------------------------
// Kernel 1: qkv = x @ W_attn + b_attn -> Q[B,H,S,64], K[B,H,S,64], Vt[B,H,64,S]
// block tile: M=256 x N=64, 8 waves; each wave: 32 rows x 64 cols (8 WMMA/chunk)
// ---------------------------------------------------------------------------
__global__ __launch_bounds__(256) void qkv_gemm_kernel(
    const float* __restrict__ x, const float* __restrict__ W,
    const float* __restrict__ bias,
    __bf16* __restrict__ Qw, __bf16* __restrict__ Kw, __bf16* __restrict__ Vw)
{
    __shared__ __attribute__((aligned(16))) __bf16 lds_a[256 * 32];  // [M][K]
    __shared__ __attribute__((aligned(16))) __bf16 lds_bt[64 * 32];  // [N][K]

    const int tid  = threadIdx.x;
    const int m0   = blockIdx.y * 256;
    const int n0   = blockIdx.x * 64;
    const int w    = tid >> 5;
    const int lane = tid & 31;
    const int ln   = lane & 15;
    const int half = lane >> 4;

    v8f acc[2][4] = {{{}, {}, {}, {}}, {{}, {}, {}, {}}};

    for (int kc = 0; kc < DIM; kc += 32) {
        // stage A: 256x32 f32 -> bf16, vector b128 loads, packed b64 LDS stores
#pragma unroll
        for (int j = 0; j < 8; ++j) {
            const int idx = tid + j * 256;              // 0..2047 (4 floats each)
            const int r = idx >> 3, c4 = (idx & 7) * 4;
            const float4 f = *(const float4*)&x[(size_t)(m0 + r) * DIM + kc + c4];
            union { unsigned long long u; unsigned short s[4]; } p;
            p.s[0] = bfbits(f.x); p.s[1] = bfbits(f.y);
            p.s[2] = bfbits(f.z); p.s[3] = bfbits(f.w);
            *(unsigned long long*)&lds_a[r * 32 + c4] = p.u;
        }
        // stage B transposed: [32 k][64 n] -> lds_bt[n][k]
#pragma unroll
        for (int j = 0; j < 2; ++j) {
            const int idx = tid + j * 256;              // 0..511 (4 floats each)
            const int kr = idx >> 4, nc4 = (idx & 15) * 4;
            const float4 f = *(const float4*)&W[(size_t)(kc + kr) * D3 + n0 + nc4];
            lds_bt[(nc4 + 0) * 32 + kr] = f2bf(f.x);
            lds_bt[(nc4 + 1) * 32 + kr] = f2bf(f.y);
            lds_bt[(nc4 + 2) * 32 + kr] = f2bf(f.z);
            lds_bt[(nc4 + 3) * 32 + kr] = f2bf(f.w);
        }
        __syncthreads();

        const int am = w * 32 + ln;
        const v16bf A0 = mk16(*(const v8bf*)&lds_a[am * 32 + 16 * half],
                              *(const v8bf*)&lds_a[am * 32 + 16 * half + 8]);
        const v16bf A1 = mk16(*(const v8bf*)&lds_a[(am + 16) * 32 + 16 * half],
                              *(const v8bf*)&lds_a[(am + 16) * 32 + 16 * half + 8]);
        v16bf Bf[4];
#pragma unroll
        for (int t = 0; t < 4; ++t) {
            const int n = t * 16 + ln;
            Bf[t] = mk16(*(const v8bf*)&lds_bt[n * 32 + 16 * half],
                         *(const v8bf*)&lds_bt[n * 32 + 16 * half + 8]);
        }
#pragma unroll
        for (int t = 0; t < 4; ++t) {
            acc[0][t] = wmma_bf16(A0, Bf[t], acc[0][t]);
            acc[1][t] = wmma_bf16(A1, Bf[t], acc[1][t]);
        }
        __syncthreads();
    }

    // epilogue: route columns to Q / K / Vt (branch is block-uniform in n0)
#pragma unroll
    for (int t = 0; t < 4; ++t) {
        const int c = n0 + t * 16 + ln;
        const float bv = bias[c];
#pragma unroll
        for (int p = 0; p < 2; ++p)
#pragma unroll
            for (int r = 0; r < 8; ++r) {
                const int g  = m0 + w * 32 + p * 16 + r + 8 * half;
                const int bb = g >> 11;
                const int sq = g & 2047;
                const float v = acc[p][t][r] + bv;
                if (n0 < DIM) {
                    const int h = c >> 6, hd = c & 63;
                    Qw[(((size_t)(bb * NH + h)) * SQ + sq) * HD + hd] = f2bf(v);
                } else if (n0 < 2 * DIM) {
                    const int c2 = c - DIM;
                    const int h = c2 >> 6, hd = c2 & 63;
                    Kw[(((size_t)(bb * NH + h)) * SQ + sq) * HD + hd] = f2bf(v);
                } else {
                    const int c2 = c - 2 * DIM;
                    const int h = c2 >> 6, hd = c2 & 63;
                    Vw[(((size_t)(bb * NH + h)) * HD + hd) * SQ + sq] = f2bf(v);
                }
            }
    }
}

// ---------------------------------------------------------------------------
// Kernel 2: causal flash attention. One wave per 16-query tile.
// ---------------------------------------------------------------------------
__global__ __launch_bounds__(256) void attn_kernel(
    const __bf16* __restrict__ Qw, const __bf16* __restrict__ Kw,
    const __bf16* __restrict__ Vw, __bf16* __restrict__ Yw)
{
    __shared__ __attribute__((aligned(16))) __bf16 pbuf[8][16 * 32];

    const int tid  = threadIdx.x;
    const int w    = tid >> 5;
    const int lane = tid & 31;
    const int ln   = lane & 15;
    const int half = lane >> 4;

    const int gw = blockIdx.x * 8 + w;
    const int bh = gw >> 7;
    const int q0 = (gw & 127) << 4;
    const int bb = bh >> 4;
    const int h  = bh & 15;

    const __bf16* Qp = Qw + (size_t)bh * SQ * HD;
    const __bf16* Kp = Kw + (size_t)bh * SQ * HD;
    const __bf16* Vp = Vw + (size_t)bh * HD * SQ;
    __bf16* pb = &pbuf[w][0];

    const __bf16* qp = Qp + (size_t)(q0 + ln) * HD + 16 * half;
    const v16bf qa0 = mk16(*(const v8bf*)(qp),      *(const v8bf*)(qp + 8));
    const v16bf qa1 = mk16(*(const v8bf*)(qp + 32), *(const v8bf*)(qp + 40));

    float mrow[8], lrow[8];
#pragma unroll
    for (int r = 0; r < 8; ++r) { mrow[r] = -1e30f; lrow[r] = 0.0f; }
    v8f acc[4] = {{}, {}, {}, {}};

    for (int k0 = 0; k0 < q0 + 16; k0 += 32) {
        v8f s0 = {}, s1 = {};
        {
            const __bf16* kp0 = Kp + (size_t)(k0 + ln) * HD + 16 * half;
            s0 = wmma_bf16(qa0, mk16(*(const v8bf*)kp0,        *(const v8bf*)(kp0 + 8)),  s0);
            s0 = wmma_bf16(qa1, mk16(*(const v8bf*)(kp0 + 32), *(const v8bf*)(kp0 + 40)), s0);
            const __bf16* kp1 = Kp + (size_t)(k0 + 16 + ln) * HD + 16 * half;
            s1 = wmma_bf16(qa0, mk16(*(const v8bf*)kp1,        *(const v8bf*)(kp1 + 8)),  s1);
            s1 = wmma_bf16(qa1, mk16(*(const v8bf*)(kp1 + 32), *(const v8bf*)(kp1 + 40)), s1);
        }

#pragma unroll
        for (int r = 0; r < 8; ++r) {
            const int qrow = q0 + r + 8 * half;
            float a0 = s0[r] * 0.125f;
            float a1 = s1[r] * 0.125f;
            if (k0 + ln > qrow)      a0 = -1e30f;
            if (k0 + 16 + ln > qrow) a1 = -1e30f;
            float tm = rmax16(fmaxf(a0, a1));
            const float mn    = fmaxf(mrow[r], tm);
            const float alpha = __expf(mrow[r] - mn);
            const float p0 = __expf(a0 - mn);
            const float p1 = __expf(a1 - mn);
            lrow[r] = lrow[r] * alpha + rsum16(p0 + p1);
            mrow[r] = mn;
#pragma unroll
            for (int t = 0; t < 4; ++t) acc[t][r] *= alpha;
            pb[(r + 8 * half) * 32 + ln]      = f2bf(p0);
            pb[(r + 8 * half) * 32 + 16 + ln] = f2bf(p1);
        }

        const v16bf pa = mk16(*(const v8bf*)&pb[ln * 32 + 16 * half],
                              *(const v8bf*)&pb[ln * 32 + 16 * half + 8]);
#pragma unroll
        for (int t = 0; t < 4; ++t) {
            const __bf16* vp = Vp + (size_t)(t * 16 + ln) * SQ + k0 + 16 * half;
            acc[t] = wmma_bf16(pa, mk16(*(const v8bf*)vp, *(const v8bf*)(vp + 8)), acc[t]);
        }
    }

    float inv[8];
#pragma unroll
    for (int r = 0; r < 8; ++r) inv[r] = 1.0f / lrow[r];
#pragma unroll
    for (int t = 0; t < 4; ++t)
#pragma unroll
        for (int r = 0; r < 8; ++r) {
            const int g = bb * SQ + q0 + r + 8 * half;
            Yw[(size_t)g * DIM + h * HD + t * 16 + ln] = f2bf(acc[t][r] * inv[r]);
        }
}

// ---------------------------------------------------------------------------
// Kernel 3: out = Y(bf16) @ W_proj + b_proj   (fp32 output)
// A-tile staged with CDNA5 async global->LDS DMA (ASYNCcnt), B-tile converted.
// block tile: M=256 x N=64, 8 waves; each wave 32x64.
// ---------------------------------------------------------------------------
__global__ __launch_bounds__(256) void proj_gemm_kernel(
    const __bf16* __restrict__ Y, const float* __restrict__ W,
    const float* __restrict__ bias, float* __restrict__ out)
{
    __shared__ __attribute__((aligned(16))) __bf16 lds_a[256 * 32];
    __shared__ __attribute__((aligned(16))) __bf16 lds_bt[64 * 32];

    const int tid  = threadIdx.x;
    const int m0   = blockIdx.y * 256;
    const int n0   = blockIdx.x * 64;
    const int w    = tid >> 5;
    const int lane = tid & 31;
    const int ln   = lane & 15;
    const int half = lane >> 4;

    v8f acc[2][4] = {{{}, {}, {}, {}}, {{}, {}, {}, {}}};

    for (int kc = 0; kc < DIM; kc += 32) {
        // stage A: bf16 -> bf16 byte copy via async global->LDS b128 DMA
#pragma unroll
        for (int j = 0; j < 4; ++j) {
            const int idx = tid + j * 256;               // 0..1023 (8 bf16 each)
            const int e = idx * 8;
            const int r = e >> 5, c = e & 31;
            const unsigned la = (unsigned)(size_t)&lds_a[e];
            const unsigned long long ga =
                (unsigned long long)(size_t)(Y + (size_t)(m0 + r) * DIM + kc + c);
            asm volatile("global_load_async_to_lds_b128 %0, %1, off"
                         :: "v"(la), "v"(ga) : "memory");
        }
        // stage B transposed with conversion
#pragma unroll
        for (int j = 0; j < 2; ++j) {
            const int idx = tid + j * 256;
            const int kr = idx >> 4, nc4 = (idx & 15) * 4;
            const float4 f = *(const float4*)&W[(size_t)(kc + kr) * DIM + n0 + nc4];
            lds_bt[(nc4 + 0) * 32 + kr] = f2bf(f.x);
            lds_bt[(nc4 + 1) * 32 + kr] = f2bf(f.y);
            lds_bt[(nc4 + 2) * 32 + kr] = f2bf(f.z);
            lds_bt[(nc4 + 3) * 32 + kr] = f2bf(f.w);
        }
        asm volatile("s_wait_asynccnt 0x0" ::: "memory");
        __syncthreads();

        const int am = w * 32 + ln;
        const v16bf A0 = mk16(*(const v8bf*)&lds_a[am * 32 + 16 * half],
                              *(const v8bf*)&lds_a[am * 32 + 16 * half + 8]);
        const v16bf A1 = mk16(*(const v8bf*)&lds_a[(am + 16) * 32 + 16 * half],
                              *(const v8bf*)&lds_a[(am + 16) * 32 + 16 * half + 8]);
        v16bf Bf[4];
#pragma unroll
        for (int t = 0; t < 4; ++t) {
            const int n = t * 16 + ln;
            Bf[t] = mk16(*(const v8bf*)&lds_bt[n * 32 + 16 * half],
                         *(const v8bf*)&lds_bt[n * 32 + 16 * half + 8]);
        }
#pragma unroll
        for (int t = 0; t < 4; ++t) {
            acc[0][t] = wmma_bf16(A0, Bf[t], acc[0][t]);
            acc[1][t] = wmma_bf16(A1, Bf[t], acc[1][t]);
        }
        __syncthreads();
    }

#pragma unroll
    for (int t = 0; t < 4; ++t) {
        const int c = n0 + t * 16 + ln;
        const float bv = bias[c];
#pragma unroll
        for (int p = 0; p < 2; ++p)
#pragma unroll
            for (int r = 0; r < 8; ++r) {
                const int g = m0 + w * 32 + p * 16 + r + 8 * half;
                out[(size_t)g * DIM + c] = acc[p][t][r] + bv;
            }
    }
}

// ---------------------------------------------------------------------------
extern "C" void kernel_launch(void* const* d_in, const int* in_sizes, int n_in,
                              void* d_out, int out_size, void* d_ws, size_t ws_size,
                              hipStream_t stream)
{
    (void)in_sizes; (void)n_in; (void)out_size; (void)ws_size;
    const float* x  = (const float*)d_in[0];
    const float* Wa = (const float*)d_in[1];
    const float* ba = (const float*)d_in[2];
    const float* Wp = (const float*)d_in[3];
    const float* bp = (const float*)d_in[4];
    float* out = (float*)d_out;

    const size_t TENS = (size_t)2 * NH * SQ * HD;   // 4,194,304 elems
    __bf16* Qw = (__bf16*)d_ws;
    __bf16* Kw = Qw + TENS;
    __bf16* Vw = Kw + TENS;
    __bf16* Yw = Vw + TENS;

    qkv_gemm_kernel<<<dim3(D3 / 64, ROWS / 256), 256, 0, stream>>>(x, Wa, ba, Qw, Kw, Vw);
    attn_kernel   <<<dim3((2 * NH * (SQ / 16)) / 8), 256, 0, stream>>>(Qw, Kw, Vw, Yw);
    proj_gemm_kernel<<<dim3(DIM / 64, ROWS / 256), 256, 0, stream>>>(Yw, Wp, bp, out);
}
